// GraphSAGENetwork_71322226917532
// MI455X (gfx1250) — matrix operand order repaired and checked by reading, not compiled, not run on hardware
//
#include <hip/hip_runtime.h>
#include <hip/hip_bf16.h>
#include <math.h>

typedef __attribute__((ext_vector_type(16))) __bf16 bf16x16;
typedef __attribute__((ext_vector_type(8)))  __bf16 bf16x8;
typedef __attribute__((ext_vector_type(8)))  float  f32x8;

#define NN 50000
#define NE 800000
#define FIN 128
#define FH 256

// ---------------- workspace layout (bytes, all 256-aligned) ----------------
static constexpr size_t O_CNT  = 0;                         // NN f32
static constexpr size_t O_AGG1 = 200192;                    // NN*128 f32
static constexpr size_t O_XIN1 = O_AGG1 + 25600000;         // NN*256 bf16
static constexpr size_t O_H1   = O_XIN1 + 25600000;         // NN*256 f32
static constexpr size_t O_AGG2 = O_H1   + 51200000;         // NN*256 f32
static constexpr size_t O_XIN2 = O_AGG2 + 51200000;         // NN*512 bf16
static constexpr size_t O_H2   = O_XIN2 + 51200000;         // NN*256 bf16
static constexpr size_t O_T    = O_H2   + 25600000;         // NN*128 bf16
static constexpr size_t O_WT1  = O_T    + 12800000;         // 256*256 bf16
static constexpr size_t O_WT2  = O_WT1  + 131072;           // 256*512 bf16
static constexpr size_t O_WTH1 = O_WT2  + 262144;           // 128*256 bf16

// ---------------------------------------------------------------------------
__global__ void sage_zero_f32(float* p, int n) {
    int i = blockIdx.x * blockDim.x + threadIdx.x;
    if (i < n) p[i] = 0.0f;
}

__global__ void sage_count_edges(const int* __restrict__ dst, float* cnt) {
    int e = blockIdx.x * blockDim.x + threadIdx.x;
    if (e < NE) atomicAdd(&cnt[dst[e]], 1.0f);
}

__global__ void sage_inv_counts(float* cnt) {
    int i = blockIdx.x * blockDim.x + threadIdx.x;
    if (i < NN) cnt[i] = 1.0f / fmaxf(cnt[i], 1.0f);
}

// thread = (edge, 4 consecutive features). A wave covers one edge's 128 feats
// contiguously -> coalesced loads, and atomic adds land in L2 (agg fits).
__global__ void sage_scatter_add(const float* __restrict__ feat,
                                 const int* __restrict__ src,
                                 const int* __restrict__ dst,
                                 float* agg, int F) {
    long t = (long)blockIdx.x * blockDim.x + threadIdx.x;
    int per_edge = F >> 2;
    long total = (long)NE * per_edge;
    if (t >= total) return;
    int e  = (int)(t / per_edge);
    int f4 = (int)(t % per_edge) << 2;
    int s = src[e], d = dst[e];
    const float4 v = *(const float4*)(feat + (size_t)s * F + f4);
    float* p = agg + (size_t)d * F + f4;
    atomicAdd(p + 0, v.x); atomicAdd(p + 1, v.y);
    atomicAdd(p + 2, v.z); atomicAdd(p + 3, v.w);
}

// xin[i] = [ agg[i]*inv[i] , right[i] ]  in bf16, row-major, K = Fl+Fr
__global__ void sage_build_xin(const float* __restrict__ agg,
                               const float* __restrict__ inv,
                               const float* __restrict__ right,
                               __bf16* __restrict__ xin, int Fl, int Fr) {
    int F = Fl + Fr;
    long t = (long)blockIdx.x * blockDim.x + threadIdx.x;
    if (t >= (long)NN * F) return;
    int i = (int)(t / F), c = (int)(t % F);
    float v = (c < Fl) ? agg[(size_t)i * Fl + c] * inv[i]
                       : right[(size_t)i * Fr + (c - Fl)];
    xin[t] = (__bf16)v;
}

// Wt[n][k] = (k < K/2 ? Wl[k][n] : Wr[k-K/2][n])  (or Wl only when Wr==null)
__global__ void sage_pack_w(const float* __restrict__ Wl,
                            const float* __restrict__ Wr,
                            __bf16* __restrict__ Wt, int K, int Nw) {
    long t = (long)blockIdx.x * blockDim.x + threadIdx.x;
    if (t >= (long)Nw * K) return;
    int n = (int)(t / K), k = (int)(t % K);
    float v;
    if (Wr) {
        int Kh = K >> 1;
        v = (k < Kh) ? Wl[(size_t)k * Nw + n] : Wr[(size_t)(k - Kh) * Nw + n];
    } else {
        v = Wl[(size_t)k * Nw + n];
    }
    Wt[t] = (__bf16)v;
}

// O = relu(X @ Wt^T + bias);  X: [NN][KTOT] bf16 row-major, Wt: [Ntiles*16][KTOT]
// One wave per 16x16 output tile; 8 waves per block. Tile index is wave-uniform
// so EXEC is all-1s at every WMMA.
template <int KTOT, bool STORE_BF16>
__global__ void sage_gemm_wmma(const __bf16* __restrict__ X,
                               const __bf16* __restrict__ Wt,
                               const float* __restrict__ bias,
                               float* __restrict__ Of32,
                               __bf16* __restrict__ Obf, int Ntiles) {
    const int lane = threadIdx.x & 31;
    const int wave = threadIdx.x >> 5;
    const int tile = blockIdx.x * (blockDim.x >> 5) + wave;
    const int Mtiles = NN / 16;
    if (tile >= Mtiles * Ntiles) return;
    const int rt = tile / Ntiles, ct = tile % Ntiles;
    const int Nw = Ntiles * 16;
    const int half = lane >> 4, l16 = lane & 15;
    const int n = ct * 16 + l16;

    f32x8 c;
    {
        float b = bias[n];
#pragma unroll
        for (int r = 0; r < 8; r++) c[r] = b;
    }

    // A: lane<16 row=l16 K={0..7,16..23}; lane>=16 same row K={8..15,24..31}
    const __bf16* arow = X  + (size_t)(rt * 16 + l16) * KTOT + half * 8;
    // B: lane<16 col=n K={0..7,16..23}; lane>=16 col=n K={8..15,24..31}
    const __bf16* brow = Wt + (size_t)n * KTOT + half * 8;

#pragma unroll
    for (int kt = 0; kt < KTOT / 32; kt++) {
        union { bf16x16 v; bf16x8 h[2]; } A, B;
        A.h[0] = *(const bf16x8*)(arow + kt * 32);
        A.h[1] = *(const bf16x8*)(arow + kt * 32 + 16);
        B.h[0] = *(const bf16x8*)(brow + kt * 32);
        B.h[1] = *(const bf16x8*)(brow + kt * 32 + 16);
        c = __builtin_amdgcn_wmma_f32_16x16x32_bf16(
                false, A.v, false, B.v, (short)0, c, false, false);
    }

    const int m0 = rt * 16 + half * 8;
#pragma unroll
    for (int r = 0; r < 8; r++) {
        float v = fmaxf(c[r], 0.0f);
        size_t off = (size_t)(m0 + r) * Nw + n;
        if (STORE_BF16) Obf[off] = (__bf16)v;
        else            Of32[off] = v;
    }
}

// out[i] = sigmoid( dot(T[i], Wh2) + bh2 )
__global__ void sage_head_out(const __bf16* __restrict__ T,
                              const float* __restrict__ Wh2,
                              const float* __restrict__ bh2,
                              float* __restrict__ out) {
    int i = blockIdx.x * blockDim.x + threadIdx.x;
    if (i >= NN) return;
    const __bf16* row = T + (size_t)i * 128;
    float acc = bh2[0];
#pragma unroll
    for (int k = 0; k < 128; k++) acc += (float)row[k] * Wh2[k];
    out[i] = 1.0f / (1.0f + __expf(-acc));
}

// ---------------------------------------------------------------------------
extern "C" void kernel_launch(void* const* d_in, const int* in_sizes, int n_in,
                              void* d_out, int out_size, void* d_ws, size_t ws_size,
                              hipStream_t stream) {
    (void)in_sizes; (void)n_in; (void)out_size; (void)ws_size;
    const float* x    = (const float*)d_in[0];
    const int*   ei   = (const int*)d_in[1];
    const int*   src  = ei;
    const int*   dst  = ei + NE;
    const float* W1l  = (const float*)d_in[2];
    const float* b1l  = (const float*)d_in[3];
    const float* W1r  = (const float*)d_in[4];
    const float* W2l  = (const float*)d_in[5];
    const float* b2l  = (const float*)d_in[6];
    const float* W2r  = (const float*)d_in[7];
    const float* Wh1  = (const float*)d_in[8];
    const float* bh1  = (const float*)d_in[9];
    const float* Wh2  = (const float*)d_in[10];
    const float* bh2  = (const float*)d_in[11];
    float* out = (float*)d_out;

    char* ws = (char*)d_ws;
    float*  cnt  = (float*)(ws + O_CNT);
    float*  agg1 = (float*)(ws + O_AGG1);
    __bf16* xin1 = (__bf16*)(ws + O_XIN1);
    float*  h1   = (float*)(ws + O_H1);
    float*  agg2 = (float*)(ws + O_AGG2);
    __bf16* xin2 = (__bf16*)(ws + O_XIN2);
    __bf16* h2   = (__bf16*)(ws + O_H2);
    __bf16* tbf  = (__bf16*)(ws + O_T);
    __bf16* wt1  = (__bf16*)(ws + O_WT1);
    __bf16* wt2  = (__bf16*)(ws + O_WT2);
    __bf16* wth1 = (__bf16*)(ws + O_WTH1);

    const int B = 256;
    auto g = [](long n, int b) { return (int)((n + b - 1) / b); };

    // zero accumulation buffers (must re-zero every call)
    sage_zero_f32<<<g(NN, B), B, 0, stream>>>(cnt, NN);
    sage_zero_f32<<<g((long)NN * FIN, B), B, 0, stream>>>(agg1, NN * FIN);
    sage_zero_f32<<<g((long)NN * FH, B), B, 0, stream>>>(agg2, NN * FH);

    // degree counts -> inverse
    sage_count_edges<<<g(NE, B), B, 0, stream>>>(dst, cnt);
    sage_inv_counts<<<g(NN, B), B, 0, stream>>>(cnt);

    // pack weights (transposed, lin_l ‖ lin_r concatenated along K) to bf16
    sage_pack_w<<<g(256L * 256, B), B, 0, stream>>>(W1l, W1r, wt1, 256, 256);
    sage_pack_w<<<g(256L * 512, B), B, 0, stream>>>(W2l, W2r, wt2, 512, 256);
    sage_pack_w<<<g(128L * 256, B), B, 0, stream>>>(Wh1, (const float*)nullptr, wth1, 256, 128);

    // ---- conv1 ----
    sage_scatter_add<<<g((long)NE * (FIN / 4), B), B, 0, stream>>>(x, src, dst, agg1, FIN);
    sage_build_xin<<<g((long)NN * 256, B), B, 0, stream>>>(agg1, cnt, x, xin1, FIN, FIN);
    sage_gemm_wmma<256, false><<<(NN / 16) * 16 / 8, 256, 0, stream>>>(
        xin1, wt1, b1l, h1, (__bf16*)nullptr, 16);

    // ---- conv2 ----
    sage_scatter_add<<<g((long)NE * (FH / 4), B), B, 0, stream>>>(h1, src, dst, agg2, FH);
    sage_build_xin<<<g((long)NN * 512, B), B, 0, stream>>>(agg2, cnt, h1, xin2, FH, FH);
    sage_gemm_wmma<512, true><<<(NN / 16) * 16 / 8, 256, 0, stream>>>(
        xin2, wt2, b2l, (float*)nullptr, h2, 16);

    // ---- head ----
    sage_gemm_wmma<256, true><<<(NN / 16) * 8 / 8, 256, 0, stream>>>(
        h2, wth1, bh1, (float*)nullptr, tbf, 8);
    sage_head_out<<<g(NN, B), B, 0, stream>>>(tbf, Wh2, bh2, out);
}
